// PointTransfomerDecModule_2680059592823
// MI455X (gfx1250) — matrix operand order repaired and checked by reading, not compiled, not run on hardware
//
#include <hip/hip_runtime.h>
#include <hip/hip_bf16.h>
#include <math.h>

typedef __attribute__((ext_vector_type(16))) _Float16 v16h;
typedef __attribute__((ext_vector_type(8)))  float    v8f;

// 1/sqrt(1 + 1e-5): eval-mode BN with running stats (0,1)
#define BN_RS 0.9999950000374997f

constexpr int B_  = 2;
constexpr int N_  = 2048;
constexpr int M_  = 8192;
constexpr int C1_ = 256;
constexpr int C_  = 128;
constexpr int K_  = 16;   // nsample
constexpr int CS_ = 16;   // C / share_planes

// ---------------------------------------------------------------------------
// Generic fused GEMM: out[m,o] = epi( sum_c X[m,c] * W[o,c] )   (Cout == 128)
//   XCHM:  X stored channel-major X[c*ldx + m]; else row-major X[m*ldx + c]
//   epi:   t = acc [*gamma[o]*BN_RS] [+beta[o]]; RELU1?; +add[m,o]?; RELU2?
//   OUT_CHM: out[o*outLdm + m] (final [B,C,M] transpose) else out[m*128+o]
// 256 threads = 8 waves; wave w owns rows [w*16, w*16+16) of a 128-row tile,
// all 8 16-col tiles (8 v8f accumulators). K stepped by 32 via ping-pong LDS
// f16 tiles with register staging: one barrier per K-step, global loads of the
// next tile overlap the current tile's WMMAs.
// ---------------------------------------------------------------------------
template<bool XCHM, bool HAS_GAMMA, bool HAS_BETA, bool RELU1, bool HAS_ADD,
         bool RELU2, bool OUT_CHM>
__global__ void __launch_bounds__(256) gemm128(
    const float* __restrict__ X, long xBatch, int ldx,
    const float* __restrict__ W,                       // [128, Kdim] row-major
    const float* __restrict__ gamma,
    const float* __restrict__ beta,
    const float* __restrict__ add, long addBatch,      // row-major [Mrows,128]
    float* __restrict__ out, long outBatch, int outLdm,
    int Mrows, int Kdim)
{
    __shared__ _Float16 Xs[2][128][40];   // 32 K + 8 pad halves
    __shared__ _Float16 Ws[2][128][40];

    const int b    = blockIdx.y;
    const int m0   = blockIdx.x * 128;
    const int tid  = threadIdx.x;
    const int lane = tid & 31;
    const int wid  = tid >> 5;
    const float* Xb = X + (long)b * xBatch;

    v8f acc[8];
    #pragma unroll
    for (int i = 0; i < 8; ++i) {
        #pragma unroll
        for (int e = 0; e < 8; ++e) acc[i][e] = 0.0f;
    }

    float xst[16], wst[16];   // register staging: 8 f16-pairs each

    auto loadX = [&](int k0) {
        #pragma unroll
        for (int i = 0; i < 8; ++i) {
            const int idx2 = i * 256 + tid;            // 2048 dword slots
            if (XCHM) {
                const int ml = idx2 & 127, cp = (idx2 >> 7) << 1;
                xst[2*i]   = Xb[(long)(k0 + cp)     * ldx + (m0 + ml)];
                xst[2*i+1] = Xb[(long)(k0 + cp + 1) * ldx + (m0 + ml)];
            } else {
                const int ml = idx2 >> 4, cp = (idx2 & 15) << 1;
                xst[2*i]   = Xb[(long)(m0 + ml) * ldx + (k0 + cp)];
                xst[2*i+1] = Xb[(long)(m0 + ml) * ldx + (k0 + cp + 1)];
            }
        }
    };
    auto loadW = [&](int k0) {
        #pragma unroll
        for (int i = 0; i < 8; ++i) {
            const int idx2 = i * 256 + tid;
            const int ol = idx2 >> 4, cp = (idx2 & 15) << 1;
            wst[2*i]   = W[(long)ol * Kdim + (k0 + cp)];
            wst[2*i+1] = W[(long)ol * Kdim + (k0 + cp + 1)];
        }
    };
    auto stagePack = [&](int buf) {
        #pragma unroll
        for (int i = 0; i < 8; ++i) {
            const int idx2 = i * 256 + tid;
            int mlx, cpx;
            if (XCHM) { mlx = idx2 & 127; cpx = (idx2 >> 7) << 1; }
            else      { mlx = idx2 >> 4;  cpx = (idx2 & 15) << 1; }
            const unsigned short ux0 = __builtin_bit_cast(unsigned short, (_Float16)xst[2*i]);
            const unsigned short ux1 = __builtin_bit_cast(unsigned short, (_Float16)xst[2*i+1]);
            *reinterpret_cast<unsigned*>(&Xs[buf][mlx][cpx]) = (unsigned)ux0 | ((unsigned)ux1 << 16);
            const int olw = idx2 >> 4, cpw = (idx2 & 15) << 1;
            const unsigned short uw0 = __builtin_bit_cast(unsigned short, (_Float16)wst[2*i]);
            const unsigned short uw1 = __builtin_bit_cast(unsigned short, (_Float16)wst[2*i+1]);
            *reinterpret_cast<unsigned*>(&Ws[buf][olw][cpw]) = (unsigned)uw0 | ((unsigned)uw1 << 16);
        }
    };

    loadX(0); loadW(0);
    int buf = 0;
    for (int k0 = 0; k0 < Kdim; k0 += 32) {
        stagePack(buf);
        __syncthreads();                       // one barrier per K-step
        if (k0 + 32 < Kdim) { loadX(k0 + 32); loadW(k0 + 32); }  // overlap w/ WMMA

        // A fragment: 16x32 f16 per ISA layout
        v16h a;
        const int arow  = wid * 16 + (lane & 15);
        const int kbase = (lane >> 4) << 3;    // 0 or 8
        #pragma unroll
        for (int h = 0; h < 16; ++h) {
            const int kk = (h & 7) + ((h >> 3) << 4) + kbase;
            a[h] = Xs[buf][arow][kk];
        }
        // 8 column tiles: B fragment 32x16 f16, D = A*B + C
        const int kb = (lane >> 4) << 3;       // 0 or 8
        #pragma unroll
        for (int ct = 0; ct < 8; ++ct) {
            v16h bf;
            const int bcol = ct * 16 + (lane & 15);
            #pragma unroll
            for (int h = 0; h < 16; ++h) bf[h] = Ws[buf][bcol][(h & 7) + ((h >> 3) << 4) + kb];
            acc[ct] = __builtin_amdgcn_wmma_f32_16x16x32_f16(
                false, a, false, bf, (short)0, acc[ct], false, false);
        }
        buf ^= 1;
    }

    // fused epilogue (all policy decisions are compile-time)
    const float* addb = HAS_ADD ? (add + (long)b * addBatch) : nullptr;
    float* outb = out + (long)b * outBatch;
    const int nb   = lane & 15;
    const int mofs = (lane >> 4) << 3;
    #pragma unroll
    for (int ct = 0; ct < 8; ++ct) {
        const int o = ct * 16 + nb;
        const float sc = HAS_GAMMA ? gamma[o] * BN_RS : 1.0f;
        const float sh = HAS_BETA  ? beta[o]          : 0.0f;
        #pragma unroll
        for (int j = 0; j < 8; ++j) {
            const int m = m0 + wid * 16 + j + mofs;
            float v = acc[ct][j];
            if (HAS_GAMMA) v *= sc;
            if (HAS_BETA)  v += sh;
            if (RELU1)     v = fmaxf(v, 0.0f);
            if (HAS_ADD)   v += addb[(long)m * 128 + o];
            if (RELU2)     v = fmaxf(v, 0.0f);
            if (OUT_CHM)   outb[(long)o * outLdm + m] = v;
            else           outb[(long)m * 128 + o]    = v;
        }
    }
}

// ---------------------------------------------------------------------------
// three_nn: per target point, 3 nearest source points (LDS-tiled scan) + weights
// ---------------------------------------------------------------------------
__global__ void __launch_bounds__(256) three_nn_kernel(
    const float* __restrict__ xyz, const float* __restrict__ txyz,
    int* __restrict__ idx3, float* __restrict__ w3)
{
    __shared__ float sx[256 * 3];
    const int g = blockIdx.x * 256 + threadIdx.x;     // block-uniform batch
    const int b = g / M_;
    const float px = txyz[(long)g * 3 + 0];
    const float py = txyz[(long)g * 3 + 1];
    const float pz = txyz[(long)g * 3 + 2];
    const float* src = xyz + (long)b * N_ * 3;

    float d0 = 3.4e38f, d1 = 3.4e38f, d2 = 3.4e38f;
    int   i0 = 0, i1 = 0, i2 = 0;
    for (int c0 = 0; c0 < N_; c0 += 256) {
        __syncthreads();
        const int t = threadIdx.x;
        sx[t * 3 + 0] = src[(long)(c0 + t) * 3 + 0];
        sx[t * 3 + 1] = src[(long)(c0 + t) * 3 + 1];
        sx[t * 3 + 2] = src[(long)(c0 + t) * 3 + 2];
        __syncthreads();
        for (int j = 0; j < 256; ++j) {
            const float dx = sx[j * 3 + 0] - px;
            const float dy = sx[j * 3 + 1] - py;
            const float dz = sx[j * 3 + 2] - pz;
            const float d  = dx * dx + dy * dy + dz * dz;
            const int idx = c0 + j;
            if (d < d2) {                              // strict <: stable ties
                if (d < d1) {
                    d2 = d1; i2 = i1;
                    if (d < d0) { d1 = d0; i1 = i0; d0 = d; i0 = idx; }
                    else        { d1 = d;  i1 = idx; }
                } else { d2 = d; i2 = idx; }
            }
        }
    }
    const float r0 = 1.0f / (sqrtf(fmaxf(d0, 0.f)) + 1e-8f);
    const float r1 = 1.0f / (sqrtf(fmaxf(d1, 0.f)) + 1e-8f);
    const float r2 = 1.0f / (sqrtf(fmaxf(d2, 0.f)) + 1e-8f);
    const float inv = 1.0f / (r0 + r1 + r2);
    idx3[(long)g * 3 + 0] = i0; idx3[(long)g * 3 + 1] = i1; idx3[(long)g * 3 + 2] = i2;
    w3[(long)g * 3 + 0] = r0 * inv; w3[(long)g * 3 + 1] = r1 * inv; w3[(long)g * 3 + 2] = r2 * inv;
}

// ---------------------------------------------------------------------------
// three_interpolate: interp[b,m,c] = sum_j w_j * f[b, idx_j, c]   (f row-major)
// ---------------------------------------------------------------------------
__global__ void __launch_bounds__(256) interp_kernel(
    const float* __restrict__ f, const int* __restrict__ idx3,
    const float* __restrict__ w3, float* __restrict__ interp)
{
    const long g  = (long)blockIdx.x * 256 + threadIdx.x;  // B*M*C
    const int  c  = (int)(g & 127);
    const long bm = g >> 7;
    const int  b  = (int)(bm / M_);
    const float* fb = f + (long)b * N_ * C_;
    const int  i0 = idx3[bm * 3 + 0], i1 = idx3[bm * 3 + 1], i2 = idx3[bm * 3 + 2];
    const float w0 = w3[bm * 3 + 0], w1 = w3[bm * 3 + 1], w2 = w3[bm * 3 + 2];
    interp[g] = w0 * fb[(long)i0 * C_ + c] + w1 * fb[(long)i1 * C_ + c] + w2 * fb[(long)i2 * C_ + c];
}

// ---------------------------------------------------------------------------
// kNN (top-16) among target points: streaming register top-K, LDS-tiled points
// ---------------------------------------------------------------------------
__global__ void __launch_bounds__(256) knn_kernel(
    const float* __restrict__ txyz, int* __restrict__ knn)
{
    __shared__ float sx[256 * 3];
    const int g = blockIdx.x * 256 + threadIdx.x;
    const int b = g / M_;
    const float px = txyz[(long)g * 3 + 0];
    const float py = txyz[(long)g * 3 + 1];
    const float pz = txyz[(long)g * 3 + 2];
    const float* src = txyz + (long)b * M_ * 3;

    float dk[16]; int ik[16];
    #pragma unroll
    for (int t = 0; t < 16; ++t) { dk[t] = 3.4e38f; ik[t] = 0; }

    for (int c0 = 0; c0 < M_; c0 += 256) {
        __syncthreads();
        const int t = threadIdx.x;
        sx[t * 3 + 0] = src[(long)(c0 + t) * 3 + 0];
        sx[t * 3 + 1] = src[(long)(c0 + t) * 3 + 1];
        sx[t * 3 + 2] = src[(long)(c0 + t) * 3 + 2];
        __syncthreads();
        for (int j = 0; j < 256; ++j) {
            const float dx = sx[j * 3 + 0] - px;
            const float dy = sx[j * 3 + 1] - py;
            const float dz = sx[j * 3 + 2] - pz;
            const float d  = dx * dx + dy * dy + dz * dz;
            if (d < dk[15]) {
                float v = d; int vi = c0 + j;
                #pragma unroll
                for (int t2 = 0; t2 < 16; ++t2) {
                    if (v < dk[t2]) {
                        float tv = dk[t2]; int ti = ik[t2];
                        dk[t2] = v; ik[t2] = vi; v = tv; vi = ti;
                    }
                }
            }
        }
    }
    #pragma unroll
    for (int t = 0; t < 16; ++t) knn[(long)g * 16 + t] = ik[t];
}

// ---------------------------------------------------------------------------
// Vector attention: one 128-thread block per target point.
//   pr = posMLP(rel_xyz); r = k_nb - q + pr; w = MLP(r) -> softmax over K;
//   agg[c] = sum_k (v_nb+pr)[k][c] * w[k][c%16];  x2 = relu(bn2(agg))
// ---------------------------------------------------------------------------
__global__ void __launch_bounds__(128) attn_kernel(
    const float* __restrict__ qb, const float* __restrict__ kb, const float* __restrict__ vb,
    const int* __restrict__ knn, const float* __restrict__ txyz,
    const float* __restrict__ p1_w, const float* __restrict__ p1_b,
    const float* __restrict__ p_bn_g, const float* __restrict__ p_bn_b,
    const float* __restrict__ p2_w, const float* __restrict__ p2_b,
    const float* __restrict__ w_bn1_g, const float* __restrict__ w_bn1_b,
    const float* __restrict__ w1w, const float* __restrict__ w1b,
    const float* __restrict__ w_bn2_g, const float* __restrict__ w_bn2_b,
    const float* __restrict__ w2w, const float* __restrict__ w2b,
    const float* __restrict__ bn2_g, const float* __restrict__ bn2_b,
    float* __restrict__ x2)
{
    __shared__ float s_rr[16][132];   // relu(bn1(r))      (pad: no 64-bank clash)
    __shared__ float s_vp[16][132];   // v + pr
    __shared__ float s_p3[16][3];
    __shared__ float s_w1h[16][16];
    __shared__ float s_w2[16][16];
    __shared__ float s_sm[16][16];
    __shared__ int   s_idx[16];

    const long bm = blockIdx.x;                 // b*M + m
    const int  b  = (int)(bm / M_);
    const int  tid = threadIdx.x;

    if (tid < 16) {
        const int j = knn[bm * 16 + tid];
        s_idx[tid] = j;
        const float pcx = txyz[bm * 3 + 0], pcy = txyz[bm * 3 + 1], pcz = txyz[bm * 3 + 2];
        const float* tb = txyz + (long)b * M_ * 3;
        const float dx = tb[(long)j * 3 + 0] - pcx;
        const float dy = tb[(long)j * 3 + 1] - pcy;
        const float dz = tb[(long)j * 3 + 2] - pcz;
        #pragma unroll
        for (int i = 0; i < 3; ++i) {
            float u = p1_w[i * 3 + 0] * dx + p1_w[i * 3 + 1] * dy + p1_w[i * 3 + 2] * dz + p1_b[i];
            u = u * (p_bn_g[i] * BN_RS) + p_bn_b[i];
            s_p3[tid][i] = fmaxf(u, 0.0f);
        }
    }
    __syncthreads();

    // phase 1: per-channel r, rr, v+pr  (k/v rows are L2-resident: 8MB each)
    const int c = tid;
    const float qv   = qb[bm * 128 + c];
    const float p2w0 = p2_w[c * 3 + 0], p2w1 = p2_w[c * 3 + 1], p2w2 = p2_w[c * 3 + 2];
    const float p2bc = p2_b[c];
    const float g1   = w_bn1_g[c] * BN_RS, b1 = w_bn1_b[c];
    for (int kk = 0; kk < 16; ++kk) {
        const long row = ((long)b * M_ + s_idx[kk]) * 128;
        if (kk + 1 < 16) {                       // gfx1250 global_prefetch_b8
            const long nrow = ((long)b * M_ + s_idx[kk + 1]) * 128;
            __builtin_prefetch(kb + nrow + c, 0, 1);
            __builtin_prefetch(vb + nrow + c, 0, 1);
        }
        const float kx = kb[row + c];
        const float vx = vb[row + c];
        const float pr = p2w0 * s_p3[kk][0] + p2w1 * s_p3[kk][1] + p2w2 * s_p3[kk][2] + p2bc;
        const float r  = kx - qv + pr;
        s_rr[kk][c] = fmaxf(r * g1 + b1, 0.0f);
        s_vp[kk][c] = vx + pr;
    }
    __syncthreads();

    // phase 2: w1[k][cs] = b + sum_c rr*W1[cs][c]; then bn2+relu
    for (int it = 0; it < 2; ++it) {
        const int pair = it * 128 + tid;
        const int kk = pair >> 4, cs = pair & 15;
        float a = w1b[cs];
        const float* wr = w1w + (long)cs * 128;
        for (int cc = 0; cc < 128; ++cc) a += s_rr[kk][cc] * wr[cc];
        s_w1h[kk][cs] = fmaxf(a * (w_bn2_g[cs] * BN_RS) + w_bn2_b[cs], 0.0f);
    }
    __syncthreads();

    // phase 3: w2[k][cs]
    for (int it = 0; it < 2; ++it) {
        const int pair = it * 128 + tid;
        const int kk = pair >> 4, cs = pair & 15;
        float a = w2b[cs];
        const float* wr = w2w + (long)cs * 16;
        #pragma unroll
        for (int cc = 0; cc < 16; ++cc) a += s_w1h[kk][cc] * wr[cc];
        s_w2[kk][cs] = a;
    }
    __syncthreads();

    // phase 4: softmax over K per cs
    if (tid < 16) {
        const int cs = tid;
        float mx = s_w2[0][cs];
        for (int kk = 1; kk < 16; ++kk) mx = fmaxf(mx, s_w2[kk][cs]);
        float sum = 0.0f;
        for (int kk = 0; kk < 16; ++kk) { const float e = expf(s_w2[kk][cs] - mx); s_sm[kk][cs] = e; sum += e; }
        const float inv = 1.0f / sum;
        for (int kk = 0; kk < 16; ++kk) s_sm[kk][cs] *= inv;
    }
    __syncthreads();

    // phase 5: aggregate + fused relu(bn2)
    float a = 0.0f;
    const int cs = c & 15;
    for (int kk = 0; kk < 16; ++kk) a += s_vp[kk][c] * s_sm[kk][cs];
    x2[bm * 128 + c] = fmaxf(a * (bn2_g[c] * BN_RS) + bn2_b[c], 0.0f);
}

// ---------------------------------------------------------------------------
extern "C" void kernel_launch(void* const* d_in, const int* in_sizes, int n_in,
                              void* d_out, int out_size, void* d_ws, size_t ws_size,
                              hipStream_t stream)
{
    const float* xyz        = (const float*)d_in[0];
    const float* feature    = (const float*)d_in[1];
    const float* txyz       = (const float*)d_in[2];
    const float* tfeat      = (const float*)d_in[3];
    const float* tu_lin1_w  = (const float*)d_in[4];
    const float* tu_bn1_g   = (const float*)d_in[5];
    const float* tu_bn1_b   = (const float*)d_in[6];
    const float* tu_lin2_w  = (const float*)d_in[7];
    const float* tu_bn2_g   = (const float*)d_in[8];
    const float* tu_bn2_b   = (const float*)d_in[9];
    const float* blk_lin1_w = (const float*)d_in[10];
    const float* blk_bn1_g  = (const float*)d_in[11];
    const float* blk_bn1_b  = (const float*)d_in[12];
    const float* wq = (const float*)d_in[13]; const float* bq = (const float*)d_in[14];
    const float* wk = (const float*)d_in[15]; const float* bk = (const float*)d_in[16];
    const float* wv = (const float*)d_in[17]; const float* bv = (const float*)d_in[18];
    const float* p1_w = (const float*)d_in[19]; const float* p1_b = (const float*)d_in[20];
    const float* p_bn_g = (const float*)d_in[21]; const float* p_bn_b = (const float*)d_in[22];
    const float* p2_w = (const float*)d_in[23]; const float* p2_b = (const float*)d_in[24];
    const float* w_bn1_g = (const float*)d_in[25]; const float* w_bn1_b = (const float*)d_in[26];
    const float* w_lin1_w = (const float*)d_in[27]; const float* w_lin1_b = (const float*)d_in[28];
    const float* w_bn2_g = (const float*)d_in[29]; const float* w_bn2_b = (const float*)d_in[30];
    const float* w_lin2_w = (const float*)d_in[31]; const float* w_lin2_b = (const float*)d_in[32];
    const float* blk_bn2_g = (const float*)d_in[33]; const float* blk_bn2_b = (const float*)d_in[34];
    const float* blk_lin3_w = (const float*)d_in[35];
    const float* blk_bn3_g = (const float*)d_in[36]; const float* blk_bn3_b = (const float*)d_in[37];

    // workspace carving (floats)
    float* ws = (float*)d_ws;
    float* f      = ws;              ws += (size_t)B_ * N_ * C_;   // [B,N,128] row-major
    int*   idx3   = (int*)ws;        ws += (size_t)B_ * M_ * 3;
    float* w3     = ws;              ws += (size_t)B_ * M_ * 3;
    float* interp = ws;              ws += (size_t)B_ * M_ * C_;
    float* feat   = ws;              ws += (size_t)B_ * M_ * C_;   // identity
    float* x1     = ws;              ws += (size_t)B_ * M_ * C_;
    float* qbuf   = ws;              ws += (size_t)B_ * M_ * C_;
    float* kbuf   = ws;              ws += (size_t)B_ * M_ * C_;
    float* vbuf   = ws;              ws += (size_t)B_ * M_ * C_;
    int*   knnb   = (int*)ws;        ws += (size_t)B_ * M_ * K_;
    float* x2     = ws;              ws += (size_t)B_ * M_ * C_;

    const long MC = (long)M_ * C_;

    // 1) f = relu(bn1(lin1 @ feature))  — channel-major input, WMMA
    gemm128<true, true, true, true, false, false, false><<<dim3(N_ / 128, B_), 256, 0, stream>>>(
        feature, (long)C1_ * N_, N_, tu_lin1_w, tu_bn1_g, tu_bn1_b,
        nullptr, 0, f, (long)N_ * C_, 0, N_, C1_);

    // 2) three_nn + weights
    three_nn_kernel<<<(B_ * M_) / 256, 256, 0, stream>>>(xyz, txyz, idx3, w3);

    // 3) three_interpolate
    interp_kernel<<<(B_ * M_ * C_) / 256, 256, 0, stream>>>(f, idx3, w3, interp);

    // 4) feat = relu(bn2(lin2 @ target_feature)) + interp   (relu BEFORE add)
    gemm128<true, true, true, true, true, false, false><<<dim3(M_ / 128, B_), 256, 0, stream>>>(
        tfeat, (long)C_ * M_, M_, tu_lin2_w, tu_bn2_g, tu_bn2_b,
        interp, MC, feat, MC, 0, M_, C_);

    // 5) x1 = relu(bn(feat @ lin1.T))
    gemm128<false, true, true, true, false, false, false><<<dim3(M_ / 128, B_), 256, 0, stream>>>(
        feat, MC, C_, blk_lin1_w, blk_bn1_g, blk_bn1_b, nullptr, 0, x1, MC, 0, M_, C_);

    // 6) q / k / v  (bias only: no gamma, beta = bias)
    gemm128<false, false, true, false, false, false, false><<<dim3(M_ / 128, B_), 256, 0, stream>>>(
        x1, MC, C_, wq, nullptr, bq, nullptr, 0, qbuf, MC, 0, M_, C_);
    gemm128<false, false, true, false, false, false, false><<<dim3(M_ / 128, B_), 256, 0, stream>>>(
        x1, MC, C_, wk, nullptr, bk, nullptr, 0, kbuf, MC, 0, M_, C_);
    gemm128<false, false, true, false, false, false, false><<<dim3(M_ / 128, B_), 256, 0, stream>>>(
        x1, MC, C_, wv, nullptr, bv, nullptr, 0, vbuf, MC, 0, M_, C_);

    // 7) kNN top-16 among targets
    knn_kernel<<<(B_ * M_) / 256, 256, 0, stream>>>(txyz, knnb);

    // 8) vector attention -> x2 = relu(bn2(agg))
    attn_kernel<<<B_ * M_, 128, 0, stream>>>(
        qbuf, kbuf, vbuf, knnb, txyz,
        p1_w, p1_b, p_bn_g, p_bn_b, p2_w, p2_b,
        w_bn1_g, w_bn1_b, w_lin1_w, w_lin1_b,
        w_bn2_g, w_bn2_b, w_lin2_w, w_lin2_b,
        blk_bn2_g, blk_bn2_b, x2);

    // 9) out[b,c,m] = relu(bn3(x2 @ lin3.T) + identity)  — fused transpose
    gemm128<false, true, true, false, true, true, true><<<dim3(M_ / 128, B_), 256, 0, stream>>>(
        x2, MC, C_, blk_lin3_w, blk_bn3_g, blk_bn3_b,
        feat, MC, (float*)d_out, (long)C_ * M_, M_, M_, C_);
}
